// MultiHeadAttention_38379827757682
// MI455X (gfx1250) — matrix operand order repaired and checked by reading, not compiled, not run on hardware
//
#include <hip/hip_runtime.h>

// ---------------------------------------------------------------------------
// MI455X (gfx1250) multi-head cross-attention, causal, flash-style.
//   B=8, TX=TZ=1024, DX=DZ=1024, H=16, DATT=DMID=64, DOUT=1024
// bf16 internally (v_wmma_f32_16x16x32_bf16, f32 accum), weights
// pre-transposed so all WMMA fragments are aligned b128 loads, flash
// attention with per-wave-independent tiles (wave32, zero barriers),
// causal mask hoisted: unmasked steady-state tiles + one diagonal tile.
// Working set (~90MB) is L2-resident (192MB).
// ---------------------------------------------------------------------------

typedef __attribute__((ext_vector_type(16))) __bf16 v16bf;
typedef __attribute__((ext_vector_type(8)))  float  v8f;

static_assert(sizeof(v16bf) == 32, "v16bf must be 32 bytes");
static_assert(sizeof(v8f)   == 32, "v8f must be 32 bytes");

union FragBf { v16bf v; uint4 q[2]; };

#define DMODEL 1024   // DX = DZ = H*DATT = H*DMID = DOUT
#define NB     8      // batch
#define NT     1024   // TX = TZ
#define NH     16
#define DHEAD  64

// ---------------------------------------------------------------------------
// Pass 0: conversions / weight re-layout
// ---------------------------------------------------------------------------

// f32 -> bf16, 4 elements per thread
__global__ void k_cvt4(const float4* __restrict__ s, __bf16* __restrict__ d, int n4) {
  int i = blockIdx.x * blockDim.x + threadIdx.x;
  if (i >= n4) return;
  float4 f = s[i];
  int o = i * 4;
  d[o + 0] = (__bf16)f.x;
  d[o + 1] = (__bf16)f.y;
  d[o + 2] = (__bf16)f.z;
  d[o + 3] = (__bf16)f.w;
}

// Wq/Wk/Wv [H][D][64] f32  ->  Wt [N=H*64][K=D] bf16 (transposed, row-major K)
__global__ void k_wqkv_t(const float* __restrict__ src, __bf16* __restrict__ dst) {
  int i = blockIdx.x * blockDim.x + threadIdx.x;   // 1M threads
  int n = i >> 10, d = i & 1023;
  int h = n >> 6,  e = n & 63;
  dst[i] = (__bf16)src[(h << 16) + (d << 6) + e];  // h*1024*64 + d*64 + e
}

// Wp [K=1024][N=1024] f32 -> WpT [N][K] bf16
__global__ void k_wp_t(const float* __restrict__ src, __bf16* __restrict__ dst) {
  int i = blockIdx.x * blockDim.x + threadIdx.x;   // 1M threads
  int n = i >> 10, k = i & 1023;
  dst[i] = (__bf16)src[(k << 10) + n];
}

// ---------------------------------------------------------------------------
// GEMM: C[M,1024] = A[M,1024](bf16) @ Bt[1024,1024]^T(bf16) + bias
//   Bt stored transposed: Bt[n][k].  K = N = 1024 fixed, M = 8192.
// Block: 256 thr = 8 waves; wave tile 32x64 (2x4 wmma accums); block 128x128.
// MODE 0: f32 row-major out   MODE 1: bf16 row-major out
// MODE 2: bf16 out scattered as V-transposed  Vt[(b*H+h)*64+e][z]
// ---------------------------------------------------------------------------
template <int MODE>
__global__ __launch_bounds__(256) void k_gemm(const __bf16* __restrict__ A,
                                              const __bf16* __restrict__ Bt,
                                              const float*  __restrict__ bias,
                                              void* __restrict__ Cout) {
  const int tid  = threadIdx.x;
  const int wid  = tid >> 5;
  const int lane = tid & 31;
  const int l15  = lane & 15;
  const int hi8  = (lane & 16) ? 8  : 0;
  const int hi16 = (lane & 16) ? 16 : 0;

  const int m0 = blockIdx.x * 128 + (wid & 3) * 32;   // wave M base
  const int n0 = blockIdx.y * 128 + (wid >> 2) * 64;  // wave N base

  v8f acc[2][4];
#pragma unroll
  for (int mi = 0; mi < 2; ++mi)
#pragma unroll
    for (int ni = 0; ni < 4; ++ni)
      acc[mi][ni] = (v8f){0.f, 0.f, 0.f, 0.f, 0.f, 0.f, 0.f, 0.f};

  const __bf16* arow = A  + (size_t)(m0 + l15) * DMODEL;
  const __bf16* brow = Bt + (size_t)(n0 + l15) * DMODEL;

  for (int k = 0; k < DMODEL; k += 32) {
    FragBf af[2], bfr[4];
#pragma unroll
    for (int mi = 0; mi < 2; ++mi) {
      // A layout: lanes0-15 halves[0..7]=K+0..7, [8..15]=K+16..23; hi lanes +8
      const __bf16* p = arow + (size_t)mi * 16 * DMODEL + k + hi8;
      af[mi].q[0] = *(const uint4*)(p);
      af[mi].q[1] = *(const uint4*)(p + 16);
    }
#pragma unroll
    for (int ni = 0; ni < 4; ++ni) {
      // B layout: lane = column, halves[0..7]=K+0..7, [8..15]=K+16..23; hi lanes +8
      const __bf16* p = brow + (size_t)ni * 16 * DMODEL + k + hi8;
      bfr[ni].q[0] = *(const uint4*)(p);
      bfr[ni].q[1] = *(const uint4*)(p + 16);
    }
#pragma unroll
    for (int mi = 0; mi < 2; ++mi)
#pragma unroll
      for (int ni = 0; ni < 4; ++ni)
        acc[mi][ni] = __builtin_amdgcn_wmma_f32_16x16x32_bf16(
            false, af[mi].v, false, bfr[ni].v, (short)0, acc[mi][ni], false, false);
  }

#pragma unroll
  for (int mi = 0; mi < 2; ++mi)
#pragma unroll
    for (int ni = 0; ni < 4; ++ni) {
      const int col = n0 + ni * 16 + l15;
      const float bcol = bias[col];
#pragma unroll
      for (int r = 0; r < 8; ++r) {
        const int row = m0 + mi * 16 + hi8 + r;   // C layout: VGPR r -> M=r (+8 hi)
        const float v = acc[mi][ni][r] + bcol;
        if (MODE == 0) {
          ((float*)Cout)[(size_t)row * DMODEL + col] = v;
        } else if (MODE == 1) {
          ((__bf16*)Cout)[(size_t)row * DMODEL + col] = (__bf16)v;
        } else {
          const int bb = row >> 10, zz = row & 1023;
          const int hh = col >> 6,  ee = col & 63;
          ((__bf16*)Cout)[(((size_t)bb * NH + hh) * DHEAD + ee) * NT + zz] = (__bf16)v;
        }
      }
    }
}

// ---------------------------------------------------------------------------
// One 16(q-rows) x 64(z-cols) flash-attention tile for a single wave.
// MASKED=false: steady-state tile fully below the diagonal (no mask code).
// MASKED=true : the single diagonal tile (per-element causal mask).
// ---------------------------------------------------------------------------
template <bool MASKED>
__device__ __forceinline__ void attn_tile(
    const int z0, const int r0, const int l15, const int hi8, const int hi16,
    const __bf16* __restrict__ Kp, const __bf16* __restrict__ Vp,
    const FragBf aq[2], v8f yacc[4], float mrow[8], float lrow[8],
    __bf16 (* __restrict__ ldsW)[72]) {          // this wave's [16][72] P tile
  const float scale = 0.125f;                    // 1/sqrt(64)

  // ---- S = Q K^T for 16 rows x 64 z-cols --------------------------------
  v8f sc[4];
#pragma unroll
  for (int ni = 0; ni < 4; ++ni)
    sc[ni] = (v8f){0.f, 0.f, 0.f, 0.f, 0.f, 0.f, 0.f, 0.f};
#pragma unroll
  for (int ks = 0; ks < 2; ++ks)
#pragma unroll
    for (int ni = 0; ni < 4; ++ni) {
      const __bf16* p = Kp + (size_t)(z0 + ni * 16 + l15) * DMODEL + ks * 32 + hi8;
      FragBf bk;
      bk.q[0] = *(const uint4*)(p);
      bk.q[1] = *(const uint4*)(p + 16);
      sc[ni] = __builtin_amdgcn_wmma_f32_16x16x32_bf16(
          false, aq[ks].v, false, bk.v, (short)0, sc[ni], false, false);
    }

  // ---- scale (+ causal mask on the diagonal tile only) -------------------
#pragma unroll
  for (int ni = 0; ni < 4; ++ni) {
#pragma unroll
    for (int r = 0; r < 8; ++r) {
      float s = sc[ni][r] * scale;
      if (MASKED) {
        const int zg = z0 + ni * 16 + l15;
        const int qg = r0 + hi8 + r;
        s = (zg > qg) ? -1e30f : s;
      }
      sc[ni][r] = s;
    }
  }

  // ---- online softmax (row stats per VGPR element; 16-lane reductions) ---
  float alpha[8];
#pragma unroll
  for (int r = 0; r < 8; ++r) {
    float v = fmaxf(fmaxf(sc[0][r], sc[1][r]), fmaxf(sc[2][r], sc[3][r]));
#pragma unroll
    for (int off = 8; off > 0; off >>= 1)
      v = fmaxf(v, __shfl_xor(v, off, 32));
    const float mn = fmaxf(mrow[r], v);
    alpha[r] = __expf(mrow[r] - mn);
    mrow[r]  = mn;
  }
  float rs[8];
#pragma unroll
  for (int r = 0; r < 8; ++r) rs[r] = 0.f;
#pragma unroll
  for (int ni = 0; ni < 4; ++ni)
#pragma unroll
    for (int r = 0; r < 8; ++r) {
      const float p = __expf(sc[ni][r] - mrow[r]);   // masked -> exactly 0
      rs[r] += p;
      ldsW[hi8 + r][ni * 16 + l15] = (__bf16)p;      // C-layout scatter to LDS
    }
#pragma unroll
  for (int r = 0; r < 8; ++r) {
#pragma unroll
    for (int off = 8; off > 0; off >>= 1)
      rs[r] += __shfl_xor(rs[r], off, 32);
    lrow[r] = lrow[r] * alpha[r] + rs[r];
  }
#pragma unroll
  for (int ni = 0; ni < 4; ++ni)
#pragma unroll
    for (int r = 0; r < 8; ++r) yacc[ni][r] *= alpha[r];

  // ---- P (A-layout gather from LDS) then y += P V ------------------------
  FragBf ap[2];
  {
    const __bf16* prow = &ldsW[l15][0];
#pragma unroll
    for (int ks = 0; ks < 2; ++ks) {
      const __bf16* p = prow + ks * 32 + hi8;
      ap[ks].q[0] = *(const uint4*)(p);
      ap[ks].q[1] = *(const uint4*)(p + 16);
    }
  }
#pragma unroll
  for (int ks = 0; ks < 2; ++ks)
#pragma unroll
    for (int ni = 0; ni < 4; ++ni) {
      const __bf16* p = Vp + (size_t)(ni * 16 + l15) * NT + z0 + ks * 32 + hi8;
      FragBf bv;
      bv.q[0] = *(const uint4*)(p);
      bv.q[1] = *(const uint4*)(p + 16);
      yacc[ni] = __builtin_amdgcn_wmma_f32_16x16x32_bf16(
          false, ap[ks].v, false, bv.v, (short)0, yacc[ni], false, false);
    }
}

// ---------------------------------------------------------------------------
// Flash attention (causal). Q,K: [B,T,H*64] bf16 (head-interleaved rows).
// Vt: [B,H,64,TZ] bf16.  Y: [B,T,H*64] bf16.
// 256 thr = 8 independent waves; wave owns 16 q-rows; z-tiles of 64.
// Exactly r0/64 unmasked tiles + one diagonal (masked) tile per wave.
// ---------------------------------------------------------------------------
__global__ __launch_bounds__(256) void k_attn(const __bf16* __restrict__ Q,
                                              const __bf16* __restrict__ K,
                                              const __bf16* __restrict__ Vt,
                                              __bf16* __restrict__ Y) {
  __shared__ __align__(16) __bf16 ldsP[8][16][72];   // per-wave 16x64 P tile, padded

  const int tid  = threadIdx.x;
  const int wid  = tid >> 5;
  const int lane = tid & 31;
  const int l15  = lane & 15;
  const int hi8  = (lane & 16) ? 8  : 0;
  const int hi16 = (lane & 16) ? 16 : 0;

  const int xt = blockIdx.x & 7;        // 8 row-tiles of 128 per (b,h)
  const int bh = blockIdx.x >> 3;
  const int b  = bh >> 4;
  const int h  = bh & 15;
  const int r0 = xt * 128 + wid * 16;   // this wave's q-row base

  const __bf16* Qp = Q  + ((size_t)(b * NT + r0)) * DMODEL + h * DHEAD;
  const __bf16* Kp = K  + ((size_t)(b * NT)) * DMODEL + h * DHEAD;
  const __bf16* Vp = Vt + ((size_t)(b * NH + h)) * DHEAD * NT;
  __bf16 (*ldsW)[72] = ldsP[wid];

  // Q fragments for both datt k-steps, kept in registers for the whole z loop
  FragBf aq[2];
  {
    const __bf16* qrow = Qp + (size_t)l15 * DMODEL;
#pragma unroll
    for (int ks = 0; ks < 2; ++ks) {
      const __bf16* p = qrow + ks * 32 + hi8;
      aq[ks].q[0] = *(const uint4*)(p);
      aq[ks].q[1] = *(const uint4*)(p + 16);
    }
  }

  float mrow[8], lrow[8];
  v8f yacc[4];
#pragma unroll
  for (int r = 0; r < 8; ++r) { mrow[r] = -1e30f; lrow[r] = 0.f; }
#pragma unroll
  for (int ni = 0; ni < 4; ++ni)
    yacc[ni] = (v8f){0.f, 0.f, 0.f, 0.f, 0.f, 0.f, 0.f, 0.f};

  // Steady state: tiles entirely below the diagonal, no mask code at all.
  const int nfull = r0 >> 6;
  for (int t = 0; t < nfull; ++t)
    attn_tile<false>(t * 64, r0, l15, hi8, hi16, Kp, Vp, aq, yacc, mrow, lrow, ldsW);
  // Single diagonal tile: covers z in [nfull*64, nfull*64+64) >= r0+15.
  attn_tile<true>(nfull * 64, r0, l15, hi8, hi16, Kp, Vp, aq, yacc, mrow, lrow, ldsW);

  // ---- epilogue: y / l, write head-interleaved bf16 ----------------------
#pragma unroll
  for (int ni = 0; ni < 4; ++ni)
#pragma unroll
    for (int r = 0; r < 8; ++r) {
      const int xrow = r0 + hi8 + r;
      const float v = yacc[ni][r] * (1.0f / lrow[r]);
      Y[((size_t)(b * NT + xrow)) * DMODEL + h * DHEAD + ni * 16 + l15] = (__bf16)v;
    }
}

// ---------------------------------------------------------------------------
// Orchestration
// ---------------------------------------------------------------------------
extern "C" void kernel_launch(void* const* d_in, const int* in_sizes, int n_in,
                              void* d_out, int out_size, void* d_ws, size_t ws_size,
                              hipStream_t stream) {
  (void)in_sizes; (void)n_in; (void)out_size; (void)ws_size;
  const float* x  = (const float*)d_in[0];
  const float* z  = (const float*)d_in[1];
  const float* Wq = (const float*)d_in[2];
  const float* bq = (const float*)d_in[3];
  const float* Wk = (const float*)d_in[4];
  const float* bk = (const float*)d_in[5];
  const float* Wv = (const float*)d_in[6];
  const float* bv = (const float*)d_in[7];
  const float* Wp = (const float*)d_in[8];
  const float* bp = (const float*)d_in[9];
  // d_in[10] = mask: causal tril, implemented analytically in k_attn.

  const size_t MB = 1024 * 1024;
  char* ws = (char*)d_ws;
  __bf16* xb  = (__bf16*)(ws + 0);        // 16 MB  [B*T][1024]
  __bf16* zb  = (__bf16*)(ws + 16 * MB);  // 16 MB
  __bf16* WqT = (__bf16*)(ws + 32 * MB);  //  2 MB  [1024][1024] transposed
  __bf16* WkT = (__bf16*)(ws + 34 * MB);
  __bf16* WvT = (__bf16*)(ws + 36 * MB);
  __bf16* WpT = (__bf16*)(ws + 38 * MB);
  __bf16* Qb  = (__bf16*)(ws + 40 * MB);  // 16 MB  [B,T,H*64]
  __bf16* Kb  = (__bf16*)(ws + 56 * MB);  // 16 MB
  __bf16* Vt  = (__bf16*)(ws + 72 * MB);  // 16 MB  [B,H,64,TZ]
  __bf16* Yb  = xb;                       // reuse: xb dead after Q projection

  const int Mtot   = NB * NT;             // 8192
  const int nElem  = Mtot * DMODEL;       // 8M
  const int n4     = nElem / 4;

  // Pass 0: conversions
  k_cvt4<<<n4 / 256, 256, 0, stream>>>((const float4*)x, xb, n4);
  k_cvt4<<<n4 / 256, 256, 0, stream>>>((const float4*)z, zb, n4);
  k_wqkv_t<<<(DMODEL * DMODEL) / 256, 256, 0, stream>>>(Wq, WqT);
  k_wqkv_t<<<(DMODEL * DMODEL) / 256, 256, 0, stream>>>(Wk, WkT);
  k_wqkv_t<<<(DMODEL * DMODEL) / 256, 256, 0, stream>>>(Wv, WvT);
  k_wp_t<<<(DMODEL * DMODEL) / 256, 256, 0, stream>>>(Wp, WpT);

  // Pass 1: Q/K/V projections (heads fused into N=1024 columns)
  dim3 gg(Mtot / 128, DMODEL / 128);
  k_gemm<1><<<gg, 256, 0, stream>>>(xb, WqT, bq, (void*)Qb);
  k_gemm<1><<<gg, 256, 0, stream>>>(zb, WkT, bk, (void*)Kb);
  k_gemm<2><<<gg, 256, 0, stream>>>(zb, WvT, bv, (void*)Vt);   // V transposed

  // Pass 2: causal flash attention -> Yb (head-concat layout)
  k_attn<<<NB * NH * (NT / 128), 256, 0, stream>>>(Qb, Kb, Vt, Yb);

  // Pass 3: output projection, f32 + bias -> d_out
  k_gemm<0><<<gg, 256, 0, stream>>>(Yb, WpT, bp, d_out);
}
